// FasterRCNN_42494406427384
// MI455X (gfx1250) — matrix operand order repaired and checked by reading, not compiled, not run on hardware
//
#include <hip/hip_runtime.h>
#include <hip/hip_bf16.h>

typedef __attribute__((ext_vector_type(16))) _Float16 v16h;
typedef __attribute__((ext_vector_type(8)))  _Float16 v8h;
typedef __attribute__((ext_vector_type(2)))  _Float16 v2h;
typedef __attribute__((ext_vector_type(8)))  float    v8f;

#define NUM_CLASSES 21
#define N_PROP 300
#define POOL 7
#define FC_HIDDEN 256
#define NUM_A 9
#define HF 32
#define WF 32
#define BSZ 2
#define CFM 1024
#define NBOX (HF*WF*NUM_A)          // 9216
#define K_PATCH 768                 // 3*16*16
#define K_RPN  (CFM*9)              // 9216
#define K_FC1  (CFM*POOL*POOL)      // 50176
#define M_IMG  (BSZ*HF*WF)          // 2048
#define M_ROI  (BSZ*N_PROP)         // 600

__constant__ float c_anch[NUM_A][2] = {
  {89.6f,44.8f},{64.0f,64.0f},{44.8f,89.6f},{179.2f,89.6f},{128.0f,128.0f},
  {89.6f,179.2f},{358.4f,179.2f},{256.0f,256.0f},{179.2f,358.4f}};

// A-fragment K index for element j (=2i+e) and lane-half h (ISA 7.12.2, 16-bit A 16x32)
__device__ __forceinline__ int a_kmap(int h, int j) {
  return (j < 8) ? (h * 8 + j) : (16 + h * 8 + (j - 8));
}

__device__ __forceinline__ v16h frag_load(const _Float16* row) {
  v8h lo = *(const v8h*)(row);
  v8h hi = *(const v8h*)(row + 8);
  return __builtin_shufflevector(lo, hi, 0,1,2,3,4,5,6,7,8,9,10,11,12,13,14,15);
}

// ---------------------------------------------------------------------------
// Kernel 1: backbone 16x16/s16 conv as WMMA GEMM (fragment-major LDS staging).
// ---------------------------------------------------------------------------
__global__ void k_gemm_patch(const float* __restrict__ img,
                             const float* __restrict__ w_cnn,
                             const float* __restrict__ b_cnn,
                             float* __restrict__ fm) {
  __shared__ __align__(16) _Float16 Af[32][24];        // lane-major A fragments
  __shared__ __align__(16) _Float16 Bf[128][24];       // (wave*32+lane)-major B frags
  const int mT = blockIdx.x;            // 128 tiles of 16 rows
  const int nT = blockIdx.y;            // 16 tiles of 64 cols
  const int tid = threadIdx.x;          // 128 threads = 4 waves
  const int wave = tid >> 5, lane = tid & 31;
  const int lm = lane & 15, lh = lane >> 4;
  v8f acc = {};
  for (int k0 = 0; k0 < K_PATCH; k0 += 32) {
    // stage A: 256 half-pairs
    for (int e = tid; e < 256; e += 128) {
      int L = e >> 3, j = (e & 7) * 2;
      int m = mT * 16 + (L & 15), h = L >> 4;
      int b = m >> 10, yx = m & 1023, y = yx >> 5, x = yx & 31;
      float v[2];
#pragma unroll
      for (int t = 0; t < 2; ++t) {
        int k = k0 + a_kmap(h, j + t);
        int c = k >> 8, rem = k & 255, ky = rem >> 4, kx = rem & 15;
        v[t] = img[(((b * 3 + c) * 512) + y * 16 + ky) * 512 + x * 16 + kx];
      }
      *(v2h*)&Af[L][j] = (v2h){(_Float16)v[0], (_Float16)v[1]};
    }
    // stage B: 1024 half-pairs
    for (int e = tid; e < 1024; e += 128) {
      int row = e >> 3, j = (e & 7) * 2;
      int w = row >> 5, L = row & 31, h = L >> 4;
      int n = nT * 64 + w * 16 + (L & 15);
      const float* src = &w_cnn[n * K_PATCH + k0 + a_kmap(h, j)];
      *(v2h*)&Bf[row][j] = (v2h){(_Float16)src[0], (_Float16)src[1]};
    }
    __syncthreads();
    v16h a  = frag_load(&Af[lane][0]);
    v16h bf = frag_load(&Bf[wave * 32 + lane][0]);
    acc = __builtin_amdgcn_wmma_f32_16x16x32_f16(false, a, false, bf, (short)0, acc, false, false);
    __syncthreads();
  }
  const int n = nT * 64 + wave * 16 + lm;
  const float bias = b_cnn[n];
#pragma unroll
  for (int j = 0; j < 8; ++j) {
    int m = mT * 16 + j + 8 * lh;
    int b = m >> 10, yx = m & 1023;
    fm[(b * CFM + n) * (HF * WF) + yx] = acc[j] + bias;
  }
}

// ---------------------------------------------------------------------------
// Kernel 2: RPN 3x3 SAME conv, implicit im2col WMMA GEMM, N=45 padded to 48.
// ---------------------------------------------------------------------------
__global__ void k_gemm_rpn(const float* __restrict__ fm,
                           const float* __restrict__ w_rpn,
                           const float* __restrict__ b_rpn,
                           float* __restrict__ rpnc) {
  __shared__ __align__(16) _Float16 Af[32][24];
  __shared__ __align__(16) _Float16 Bf[96][24];
  const int mT = blockIdx.x;            // 128 tiles
  const int tid = threadIdx.x;          // 96 threads = 3 waves
  const int wave = tid >> 5, lane = tid & 31;
  const int lm = lane & 15, lh = lane >> 4;
  v8f acc = {};
  for (int k0 = 0; k0 < K_RPN; k0 += 32) {
    for (int e = tid; e < 256; e += 96) {
      int L = e >> 3, j = (e & 7) * 2;
      int m = mT * 16 + (L & 15), h = L >> 4;
      int b = m >> 10, yx = m & 1023, y = yx >> 5, x = yx & 31;
      float v[2];
#pragma unroll
      for (int t = 0; t < 2; ++t) {
        int k = k0 + a_kmap(h, j + t);
        int c = k / 9, rem = k - c * 9, dy = rem / 3, dx = rem - dy * 3;
        int iy = y + dy - 1, ix = x + dx - 1;
        v[t] = (iy >= 0 && iy < HF && ix >= 0 && ix < WF)
                 ? fm[(b * CFM + c) * (HF * WF) + iy * WF + ix] : 0.f;
      }
      *(v2h*)&Af[L][j] = (v2h){(_Float16)v[0], (_Float16)v[1]};
    }
    for (int e = tid; e < 768; e += 96) {
      int row = e >> 3, j = (e & 7) * 2;
      int w = row >> 5, L = row & 31, h = L >> 4;
      int n = w * 16 + (L & 15);
      float v0 = 0.f, v1 = 0.f;
      if (n < 45) {
        const float* src = &w_rpn[n * K_RPN + k0 + a_kmap(h, j)];
        v0 = src[0]; v1 = src[1];
      }
      *(v2h*)&Bf[row][j] = (v2h){(_Float16)v0, (_Float16)v1};
    }
    __syncthreads();
    v16h a  = frag_load(&Af[lane][0]);
    v16h bf = frag_load(&Bf[wave * 32 + lane][0]);
    acc = __builtin_amdgcn_wmma_f32_16x16x32_f16(false, a, false, bf, (short)0, acc, false, false);
    __syncthreads();
  }
  const int n = wave * 16 + lm;
  if (n < 45) {
    const float bias = b_rpn[n];
#pragma unroll
    for (int j = 0; j < 8; ++j) {
      int m = mT * 16 + j + 8 * lh;
      int b = m >> 10, yx = m & 1023;
      rpnc[(b * 45 + n) * (HF * WF) + yx] = acc[j] + bias;
    }
  }
}

// ---------------------------------------------------------------------------
// Kernel 3: anchors, rpn_reg/rpn_cls outputs, box decode, sigmoid, area
// ---------------------------------------------------------------------------
__global__ void k_rpn_post(const float* __restrict__ rpnc,
                           float* __restrict__ out_reg, float* __restrict__ out_cls,
                           float* __restrict__ out_anch,
                           float* __restrict__ boxes, float* __restrict__ scores,
                           float* __restrict__ area, float* __restrict__ clsc) {
  int idx = blockIdx.x * blockDim.x + threadIdx.x;
  if (idx >= BSZ * NBOX) return;
  int b = idx / NBOX, p = idx - b * NBOX;
  int yx = p / NUM_A, a = p - yx * NUM_A;
  int y = yx >> 5, x = yx & 31;
  float cy = (y + 0.5f) * 16.f, cx = (x + 0.5f) * 16.f;
  float hh = c_anch[a][0], ww = c_anch[a][1];
  float an[4] = {cy - hh * 0.5f, cx - ww * 0.5f, cy + hh * 0.5f, cx + ww * 0.5f};
  float bx[4];
#pragma unroll
  for (int j = 0; j < 4; ++j) {
    float rv = rpnc[(b * 45 + a * 5 + j) * (HF * WF) + yx];
    out_reg[(b * NBOX + p) * 4 + j] = rv;
    bx[j] = rv + an[j];
    boxes[(b * NBOX + p) * 4 + j] = bx[j];
    if (b == 0) out_anch[p * 4 + j] = an[j];
  }
  float cls = rpnc[(b * 45 + a * 5 + 4) * (HF * WF) + yx];
  out_cls[b * NBOX + p] = cls;
  clsc[b * NBOX + p] = cls;
  scores[b * NBOX + p] = 1.f / (1.f + expf(-cls));
  area[b * NBOX + p] = fmaxf(bx[2] - bx[0], 0.f) * fmaxf(bx[3] - bx[1], 0.f);
}

// ---------------------------------------------------------------------------
// Kernel 4: fixed-iteration NMS (300 rounds), one block per batch.
// ---------------------------------------------------------------------------
__global__ void k_nms(const float* __restrict__ boxes, const float* __restrict__ scores,
                      const float* __restrict__ area, int* __restrict__ kept,
                      int* __restrict__ cnt) {
  const int b = blockIdx.x, tid = threadIdx.x;
  const float* bx = boxes + b * NBOX * 4;
  const float* ar = area + b * NBOX;
  __shared__ float s[NBOX];
  __shared__ float rs[256];
  __shared__ int   ri[256];
  __shared__ float bj[4], aj;
  __shared__ int jsh;
  const float NEG = -__builtin_inff();
  for (int p = tid; p < NBOX; p += 256) s[p] = scores[b * NBOX + p];
  __syncthreads();
  int count = 0;
  for (int it = 0; it < N_PROP; ++it) {
    float bv = NEG; int bi = NBOX;
    for (int p = tid; p < NBOX; p += 256) {
      float v = s[p];
      if (v > bv || (v == bv && p < bi)) { bv = v; bi = p; }
    }
    rs[tid] = bv; ri[tid] = bi;
    __syncthreads();
    for (int off = 128; off > 0; off >>= 1) {
      if (tid < off) {
        float v = rs[tid + off]; int i2 = ri[tid + off];
        if (v > rs[tid] || (v == rs[tid] && i2 < ri[tid])) { rs[tid] = v; ri[tid] = i2; }
      }
      __syncthreads();
    }
    if (tid == 0) {
      float mv = rs[0]; int j = ri[0];
      if (mv > NEG) {
        jsh = j; kept[b * N_PROP + it] = j;
        bj[0] = bx[j*4]; bj[1] = bx[j*4+1]; bj[2] = bx[j*4+2]; bj[3] = bx[j*4+3];
        aj = ar[j];
      } else { jsh = -1; kept[b * N_PROP + it] = 0; }
    }
    __syncthreads();
    int j = jsh;
    if (j >= 0) {
      float b0 = bj[0], b1 = bj[1], b2 = bj[2], b3 = bj[3], a0 = aj;
      for (int p = tid; p < NBOX; p += 256) {
        if (s[p] == NEG && p != j) continue;
        float yy1 = fmaxf(b0, bx[p*4]);
        float xx1 = fmaxf(b1, bx[p*4+1]);
        float yy2 = fminf(b2, bx[p*4+2]);
        float xx2 = fminf(b3, bx[p*4+3]);
        float inter = fmaxf(yy2 - yy1, 0.f) * fmaxf(xx2 - xx1, 0.f);
        float iou = inter / (a0 + ar[p] - inter + 1e-9f);
        if (iou > 0.7f || p == j) s[p] = NEG;
      }
      count++;
    }
    __syncthreads();
  }
  if (tid == 0) cnt[b] = count;
}

// ---------------------------------------------------------------------------
// Kernel 5: stable descending rank-sort; keep the top-300 order entries.
// ---------------------------------------------------------------------------
__global__ void k_sort(const float* __restrict__ clsc, int* __restrict__ order_top) {
  int idx = blockIdx.x * blockDim.x + threadIdx.x;
  if (idx >= BSZ * NBOX) return;
  int b = idx / NBOX, i = idx - b * NBOX;
  const float* c = clsc + b * NBOX;
  float si = c[i];
  int rank = 0;
  for (int j = 0; j < NBOX; ++j) {
    float sj = c[j];
    rank += (sj > si) || (sj == si && j < i);
  }
  if (rank < N_PROP) order_top[b * N_PROP + rank] = i;
}

// ---------------------------------------------------------------------------
// Kernel 6: proposal selection, nms_reg/nms_cls outputs, ROI sample indices.
// ---------------------------------------------------------------------------
__global__ void k_select(const float* __restrict__ boxes, const float* __restrict__ clsc,
                         const int* __restrict__ kept, const int* __restrict__ cnt,
                         const int* __restrict__ order_top,
                         float* __restrict__ out_nreg, float* __restrict__ out_ncls,
                         int* __restrict__ idxbuf) {
  int idx = blockIdx.x * blockDim.x + threadIdx.x;
  if (idx >= M_ROI) return;
  int b = idx / N_PROP, pos = idx - b * N_PROP;
  int cb = cnt[b];
  int sel;
  if (pos < cb) sel = kept[b * N_PROP + pos];
  else {
    int pi = pos - cb; if (pi > N_PROP - 1) pi = N_PROP - 1; if (pi < 0) pi = 0;
    sel = order_top[b * N_PROP + pi];
  }
  float rg[4];
#pragma unroll
  for (int j = 0; j < 4; ++j) {
    rg[j] = boxes[(b * NBOX + sel) * 4 + j];
    out_nreg[idx * 4 + j] = rg[j];
  }
  out_ncls[idx] = clsc[b * NBOX + sel];
  const float R = 1.f / 16.f;
  float x1 = floorf(rg[1] * R), y1 = floorf(rg[0] * R);
  float x2 = ceilf(rg[3] * R),  y2 = ceilf(rg[2] * R);
  float w = fmaxf(x2 - x1, 1.f), h = fmaxf(y2 - y1, 1.f);
  int* ib = idxbuf + idx * 28;
#pragma unroll
  for (int g = 0; g < POOL; ++g) {
#pragma unroll
    for (int sIdx = 0; sIdx < 2; ++sIdx) {
      float off = 0.25f + 0.5f * sIdx;
      float ys = y1 + (g + off) * (h / POOL);
      float xs = x1 + (g + off) * (w / POOL);
      int yc = (int)floorf(ys); yc = yc < 0 ? 0 : (yc > HF - 1 ? HF - 1 : yc);
      int xc = (int)floorf(xs); xc = xc < 0 ? 0 : (xc > WF - 1 ? WF - 1 : xc);
      ib[g * 2 + sIdx] = yc;
      ib[14 + g * 2 + sIdx] = xc;
    }
  }
}

// ---------------------------------------------------------------------------
// Kernel 7: ROI max-pool -> f16 A-matrix [600][50176] for FC1 WMMA GEMM.
// ---------------------------------------------------------------------------
__global__ void k_roi_pool(const float* __restrict__ fm, const int* __restrict__ idxbuf,
                           _Float16* __restrict__ pooled) {
  long long gid = (long long)blockIdx.x * blockDim.x + threadIdx.x;
  if (gid >= (long long)M_ROI * K_FC1) return;
  int r = (int)(gid / K_FC1);
  int k = (int)(gid - (long long)r * K_FC1);
  int c = k / 49, q = k - c * 49, gy = q / 7, gx = q - gy * 7;
  int b = r / N_PROP;
  const int* ib = idxbuf + r * 28;
  int y0 = ib[gy * 2], y1 = ib[gy * 2 + 1];
  int x0 = ib[14 + gx * 2], x1 = ib[14 + gx * 2 + 1];
  const float* f = fm + (b * CFM + c) * (HF * WF);
  float v = fmaxf(fmaxf(f[y0 * WF + x0], f[y0 * WF + x1]),
                  fmaxf(f[y1 * WF + x0], f[y1 * WF + x1]));
  pooled[(long long)r * K_FC1 + k] = (_Float16)v;
}

// ---------------------------------------------------------------------------
// Kernel 8: FC1 GEMM (M=600 pad 608, K=50176, N=256) + bias + relu via WMMA.
// ---------------------------------------------------------------------------
__global__ void k_gemm_fc1(const _Float16* __restrict__ pooled,
                           const float* __restrict__ w_fc1,
                           const float* __restrict__ b_fc1,
                           float* __restrict__ hid) {
  __shared__ __align__(16) _Float16 Af[32][24];
  __shared__ __align__(16) _Float16 Bf[128][24];
  const int mT = blockIdx.x;            // 38 tiles
  const int nT = blockIdx.y;            // 4 tiles of 64
  const int tid = threadIdx.x;          // 128 = 4 waves
  const int wave = tid >> 5, lane = tid & 31;
  const int lm = lane & 15, lh = lane >> 4;
  v8f acc = {};
  for (int k0 = 0; k0 < K_FC1; k0 += 32) {
    for (int e = tid; e < 256; e += 128) {
      int L = e >> 3, j = (e & 7) * 2;
      int m = mT * 16 + (L & 15), h = L >> 4;
      v2h v = (v2h){(_Float16)0.f, (_Float16)0.f};
      if (m < M_ROI) {
        // adjacent j,j+1 map to adjacent K -> one aligned 4B load
        v = *(const v2h*)&pooled[(long long)m * K_FC1 + k0 + a_kmap(h, j)];
        if (k0 + 64 < K_FC1)
          __builtin_prefetch(&pooled[(long long)m * K_FC1 + k0 + 64], 0, 1);
      }
      *(v2h*)&Af[L][j] = v;
    }
    for (int e = tid; e < 1024; e += 128) {
      int row = e >> 3, j = (e & 7) * 2;
      int w = row >> 5, L = row & 31, h = L >> 4;
      int n = nT * 64 + w * 16 + (L & 15);
      int k = k0 + a_kmap(h, j);
      *(v2h*)&Bf[row][j] = (v2h){(_Float16)w_fc1[k * FC_HIDDEN + n],
                                 (_Float16)w_fc1[(k + 1) * FC_HIDDEN + n]};
    }
    __syncthreads();
    v16h a  = frag_load(&Af[lane][0]);
    v16h bf = frag_load(&Bf[wave * 32 + lane][0]);
    acc = __builtin_amdgcn_wmma_f32_16x16x32_f16(false, a, false, bf, (short)0, acc, false, false);
    __syncthreads();
  }
  const int n = nT * 64 + wave * 16 + lm;
  const float bias = b_fc1[n];
#pragma unroll
  for (int j = 0; j < 8; ++j) {
    int m = mT * 16 + j + 8 * lh;
    if (m < M_ROI) hid[m * FC_HIDDEN + n] = fmaxf(acc[j] + bias, 0.f);
  }
}

// ---------------------------------------------------------------------------
// Kernel 9: small heads (K=256), plain dot products.
// ---------------------------------------------------------------------------
__global__ void k_heads(const float* __restrict__ hid,
                        const float* __restrict__ w_reg, const float* __restrict__ b_reg,
                        const float* __restrict__ w_cls, const float* __restrict__ b_cls,
                        float* __restrict__ out_rreg, float* __restrict__ out_rcls) {
  int gid = blockIdx.x * blockDim.x + threadIdx.x;
  const int NOUT = NUM_CLASSES * 4 + NUM_CLASSES;   // 105
  if (gid >= M_ROI * NOUT) return;
  int r = gid / NOUT, o = gid - r * NOUT;
  const float* h = hid + r * FC_HIDDEN;
  float acc = 0.f;
  if (o < NUM_CLASSES * 4) {
    for (int k = 0; k < FC_HIDDEN; ++k) acc += h[k] * w_reg[k * (NUM_CLASSES * 4) + o];
    out_rreg[r * (NUM_CLASSES * 4) + o] = acc + b_reg[o];
  } else {
    int oc = o - NUM_CLASSES * 4;
    for (int k = 0; k < FC_HIDDEN; ++k) acc += h[k] * w_cls[k * NUM_CLASSES + oc];
    out_rcls[r * NUM_CLASSES + oc] = acc + b_cls[oc];
  }
}

// ---------------------------------------------------------------------------
extern "C" void kernel_launch(void* const* d_in, const int* in_sizes, int n_in,
                              void* d_out, int out_size, void* d_ws, size_t ws_size,
                              hipStream_t stream) {
  const float* img   = (const float*)d_in[0];
  const float* w_cnn = (const float*)d_in[2];
  const float* b_cnn = (const float*)d_in[3];
  const float* w_rpn = (const float*)d_in[4];
  const float* b_rpn = (const float*)d_in[5];
  const float* w_fc1 = (const float*)d_in[6];
  const float* b_fc1 = (const float*)d_in[7];
  const float* w_reg = (const float*)d_in[8];
  const float* b_reg = (const float*)d_in[9];
  const float* w_cls = (const float*)d_in[10];
  const float* b_cls = (const float*)d_in[11];

  float* out = (float*)d_out;
  float* out_rpn_reg = out;                                   // 2*9216*4
  float* out_rpn_cls = out + 73728;                           // 2*9216
  float* out_nms_reg = out + 92160;                           // 2*300*4
  float* out_nms_cls = out + 94560;                           // 2*300
  float* out_rcnn_reg = out + 95160;                          // 2*300*84
  float* out_rcnn_cls = out + 145560;                         // 2*300*21
  float* out_anchors  = out + 158160;                         // 9216*4

  char* p = (char*)d_ws;
  auto carve = [&](size_t bytes) { char* r = p; p += (bytes + 255) & ~(size_t)255; return (void*)r; };
  float*    fm        = (float*)   carve((size_t)BSZ * CFM * HF * WF * 4);
  float*    rpnc      = (float*)   carve((size_t)BSZ * 45 * HF * WF * 4);
  float*    boxes     = (float*)   carve((size_t)BSZ * NBOX * 4 * 4);
  float*    scores    = (float*)   carve((size_t)BSZ * NBOX * 4);
  float*    area      = (float*)   carve((size_t)BSZ * NBOX * 4);
  float*    clsc      = (float*)   carve((size_t)BSZ * NBOX * 4);
  int*      kept      = (int*)     carve((size_t)BSZ * N_PROP * 4);
  int*      cnt       = (int*)     carve((size_t)BSZ * 4);
  int*      order_top = (int*)     carve((size_t)BSZ * N_PROP * 4);
  int*      idxbuf    = (int*)     carve((size_t)M_ROI * 28 * 4);
  _Float16* pooled    = (_Float16*)carve((size_t)M_ROI * K_FC1 * 2);
  float*    hid       = (float*)   carve((size_t)M_ROI * FC_HIDDEN * 4);

  k_gemm_patch<<<dim3(M_IMG / 16, CFM / 64), 128, 0, stream>>>(img, w_cnn, b_cnn, fm);
  k_gemm_rpn<<<dim3(M_IMG / 16), 96, 0, stream>>>(fm, w_rpn, b_rpn, rpnc);
  k_rpn_post<<<(BSZ * NBOX + 255) / 256, 256, 0, stream>>>(
      rpnc, out_rpn_reg, out_rpn_cls, out_anchors, boxes, scores, area, clsc);
  k_nms<<<BSZ, 256, 0, stream>>>(boxes, scores, area, kept, cnt);
  k_sort<<<(BSZ * NBOX + 255) / 256, 256, 0, stream>>>(clsc, order_top);
  k_select<<<(M_ROI + 255) / 256, 256, 0, stream>>>(
      boxes, clsc, kept, cnt, order_top, out_nms_reg, out_nms_cls, idxbuf);
  {
    long long tot = (long long)M_ROI * K_FC1;
    k_roi_pool<<<(unsigned)((tot + 255) / 256), 256, 0, stream>>>(fm, idxbuf, pooled);
  }
  k_gemm_fc1<<<dim3((M_ROI + 15) / 16, FC_HIDDEN / 64), 128, 0, stream>>>(pooled, w_fc1, b_fc1, hid);
  k_heads<<<(M_ROI * 105 + 255) / 256, 256, 0, stream>>>(
      hid, w_reg, b_reg, w_cls, b_cls, out_rcnn_reg, out_rcnn_cls);

  (void)in_sizes; (void)n_in; (void)out_size; (void)ws_size;
}
